// MORSpyMaster_15350213116238
// MI455X (gfx1250) — matrix operand order repaired and checked by reading, not compiled, not run on hardware
//
#include <hip/hip_runtime.h>
#include <hip/hip_bf16.h>
#include <math.h>

#define EPSN 1e-12f

typedef __attribute__((ext_vector_type(16))) __bf16 v16bf;
typedef __attribute__((ext_vector_type(16))) short  v16s;
typedef __attribute__((ext_vector_type(8)))  short  v8s;
typedef __attribute__((ext_vector_type(8)))  float  v8f;

union BF16x16 { v16s s; v16bf b; unsigned short u[16]; };

__device__ inline unsigned short f2bf(float f){
  unsigned u = __float_as_uint(f);
  unsigned r = u + 0x7FFFu + ((u >> 16) & 1u);
  return (unsigned short)(r >> 16);
}
__device__ inline unsigned fkey(float f){   // monotonic float->uint transform
  unsigned u = __float_as_uint(f);
  return (u & 0x80000000u) ? ~u : (u | 0x80000000u);
}
__device__ inline float block_reduce_sum(float v, float* red){
  int t = threadIdx.x;
  red[t] = v; __syncthreads();
  for (int s = blockDim.x >> 1; s > 0; s >>= 1){
    if (t < s) red[t] += red[t + s];
    __syncthreads();
  }
  float r = red[0]; __syncthreads();
  return r;
}

// ---------------- weight transpose+convert: W(K,N) f32 -> Wt(Npad,Kpad) bf16 -------------
__global__ void wt_prep(const float* __restrict__ W, int K, int N,
                        unsigned short* __restrict__ Wt, int Kpad, int Npad){
  long long idx = (long long)blockIdx.x * blockDim.x + threadIdx.x;
  long long tot = (long long)Kpad * Npad;
  if (idx >= tot) return;
  int n = (int)(idx / Kpad);
  int k = (int)(idx % Kpad);
  float v = (k < K && n < N) ? W[(long long)k * N + n] : 0.0f;
  Wt[idx] = f2bf(v);
}

// ---------------- pooling + concat: X[b] = [pool(neg), assas, pool(neut), pool(pos)] -----
__global__ void pool_concat(const float* __restrict__ pos, const float* __restrict__ neg,
                            const float* __restrict__ neut, const float* __restrict__ assas,
                            float* __restrict__ X){
  __shared__ float red[256];
  int b = blockIdx.x, t = threadIdx.x;
  const float* srcs[3] = { neg + (long long)b*8*768, neut + (long long)b*7*768, pos + (long long)b*9*768 };
  const int   cnts[3]  = {8, 7, 9};
  const int   offs[3]  = {0, 1536, 2304};
  for (int g = 0; g < 3; ++g){
    float m[3]; float ss = 0.f;
    for (int q = 0; q < 3; q++){
      int d = t + q*256;
      float s = 0.f;
      for (int r = 0; r < cnts[g]; ++r) s += srcs[g][r*768 + d];
      m[q] = s / (float)cnts[g];
      ss += m[q]*m[q];
    }
    float tot = block_reduce_sum(ss, red);
    float inv = 1.0f / fmaxf(sqrtf(tot), EPSN);
    for (int q = 0; q < 3; q++) X[(long long)b*3072 + offs[g] + t + q*256] = m[q]*inv;
  }
  for (int q = 0; q < 3; q++){
    int d = t + q*256;
    X[(long long)b*3072 + 768 + d] = assas[(long long)b*768 + d];
  }
}

// ---------------- normalized target matrix Tn(64,32,768) bf16; order pos|neg|neut|assas --
__global__ void tn_prep(const float* __restrict__ pos, const float* __restrict__ neg,
                        const float* __restrict__ neut, const float* __restrict__ assas,
                        unsigned short* __restrict__ Tn){
  __shared__ float red[256];
  int b = blockIdx.x >> 5;
  int c = blockIdx.x & 31;
  int t = threadIdx.x;
  const float* src = nullptr;
  if      (c < 9)  src = pos  + ((long long)b*9 + c)*768;
  else if (c < 17) src = neg  + ((long long)b*8 + (c-9))*768;
  else if (c < 24) src = neut + ((long long)b*7 + (c-17))*768;
  else if (c == 24) src = assas + (long long)b*768;
  unsigned short* dst = Tn + ((long long)b*32 + c)*768;
  if (!src){ for (int q = 0; q < 3; q++) dst[t + q*256] = 0; return; }
  float v[3]; float ss = 0.f;
  for (int q = 0; q < 3; q++){ v[q] = src[t + q*256]; ss += v[q]*v[q]; }
  float tot = block_reduce_sum(ss, red);
  float inv = 1.0f / fmaxf(sqrtf(tot), EPSN);
  for (int q = 0; q < 3; q++) dst[t + q*256] = f2bf(v[q]*inv);
}

// ---------------- MLP GEMM: Out(64,Npad) = act(A(64,Kpad) @ Wt^T + bias) ------------------
// block=128 (4 waves); wave w -> M-tile rows 16w..16w+15; block.x -> N-tile
__global__ void gemm_mlp(const float* __restrict__ A, int Kpad,
                         const unsigned short* __restrict__ Wt,
                         const float* __restrict__ bias, int Nreal,
                         float* __restrict__ Out, int Npad, int doTanh){
  int wave = threadIdx.x >> 5;
  int lane = threadIdx.x & 31;
  int half = lane >> 4;
  int lan16 = lane & 15;
  int m0 = wave * 16;
  int n0 = blockIdx.x * 16;
  int m = m0 + lan16;
  int n = n0 + lan16;
  const float* arow = A + (long long)m * Kpad;
  const unsigned short* brow = Wt + (long long)n * Kpad;
  v8f acc = {};
  for (int k0 = 0; k0 < Kpad; k0 += 32){
    BF16x16 a;
    int base = k0 + 8*half;                       // A 16x32 bf16 layout: K-runs per lane
    float4 f0 = *(const float4*)(arow + base);
    float4 f1 = *(const float4*)(arow + base + 4);
    float4 f2 = *(const float4*)(arow + base + 16);
    float4 f3 = *(const float4*)(arow + base + 20);
    a.u[0]=f2bf(f0.x); a.u[1]=f2bf(f0.y); a.u[2]=f2bf(f0.z); a.u[3]=f2bf(f0.w);
    a.u[4]=f2bf(f1.x); a.u[5]=f2bf(f1.y); a.u[6]=f2bf(f1.z); a.u[7]=f2bf(f1.w);
    a.u[8]=f2bf(f2.x); a.u[9]=f2bf(f2.y); a.u[10]=f2bf(f2.z); a.u[11]=f2bf(f2.w);
    a.u[12]=f2bf(f3.x); a.u[13]=f2bf(f3.y); a.u[14]=f2bf(f3.z); a.u[15]=f2bf(f3.w);
    BF16x16 bv;
    *(v8s*)&bv.u[0] = *(const v8s*)(brow + k0 + 8*half);        // B 32x16: same K-runs as A
    *(v8s*)&bv.u[8] = *(const v8s*)(brow + k0 + 16 + 8*half);
    acc = __builtin_amdgcn_wmma_f32_16x16x32_bf16(false, a.b, false, bv.b, (short)0, acc, false, false);
  }
  float bval = (n < Nreal) ? bias[n] : 0.0f;
  for (int r = 0; r < 8; r++){
    int mr = m0 + r + 8*half;                     // C/D layout
    float v = acc[r] + bval;
    if (doTanh) v = tanhf(v);
    Out[(long long)mr * Npad + n] = v;
  }
}

// ---------------- normalize model_out; emit f32 (output 0) + bf16 copy --------------------
__global__ void finalize_model_out(const float* __restrict__ H4, float* __restrict__ out_model,
                                   unsigned short* __restrict__ mo_bf){
  __shared__ float red[256];
  int b = blockIdx.x, t = threadIdx.x;
  const float* src = H4 + (long long)b*768;
  float v[3]; float ss = 0.f;
  for (int q = 0; q < 3; q++){ v[q] = src[t + q*256]; ss += v[q]*v[q]; }
  float tot = block_reduce_sum(ss, red);
  float inv = 1.0f / fmaxf(sqrtf(tot), EPSN);
  for (int q = 0; q < 3; q++){
    float o = v[q]*inv;
    out_model[(long long)b*768 + t + q*256] = o;
    mo_bf[(long long)b*768 + t + q*256] = f2bf(o);
  }
}

// ---------------- sims^T(100000,64) = vocab_norm @ model_out^T ---------------------------
// block=128 (4 waves) handles 16 vocab rows; wave w -> batch tile 16w
__global__ void sims_kernel(const float* __restrict__ vocab,
                            const unsigned short* __restrict__ mo_bf,
                            float* __restrict__ simsT, int Nvocab){
  __shared__ __align__(32) unsigned short As[16][768];
  int m0 = blockIdx.x * 16;
  int wave = threadIdx.x >> 5;
  int lane = threadIdx.x & 31;
  // prefetch next slab (emits global_prefetch_b8)
  if (m0 + 16 < Nvocab){
    const float* nxt = vocab + (long long)(m0 + 16)*768;
    for (int q = threadIdx.x; q < 768; q += 128) __builtin_prefetch(nxt + (size_t)q*16, 0, 1);
  }
  // normalize 16 vocab rows into LDS as bf16
  for (int j = 0; j < 4; j++){
    int row = wave*4 + j;
    const float* src = vocab + (long long)(m0 + row)*768;
    float ss = 0.f;
    for (int d = lane; d < 768; d += 32){ float x = src[d]; ss += x*x; }
    for (int off = 16; off > 0; off >>= 1) ss += __shfl_xor(ss, off, 32);
    float inv = 1.0f / fmaxf(sqrtf(ss), EPSN);
    for (int d = lane; d < 768; d += 32) As[row][d] = f2bf(src[d]*inv);
  }
  __syncthreads();
  int half = lane >> 4, lan16 = lane & 15;
  int n0 = wave * 16;
  const unsigned short* brow = mo_bf + (long long)(n0 + lan16)*768;
  v8f acc = {};
  for (int k0 = 0; k0 < 768; k0 += 32){
    BF16x16 a;
    *(v8s*)&a.u[0] = *(const v8s*)(&As[lan16][k0 + 8*half]);
    *(v8s*)&a.u[8] = *(const v8s*)(&As[lan16][k0 + 16 + 8*half]);
    BF16x16 bv;
    *(v8s*)&bv.u[0] = *(const v8s*)(brow + k0 + 8*half);
    *(v8s*)&bv.u[8] = *(const v8s*)(brow + k0 + 16 + 8*half);
    acc = __builtin_amdgcn_wmma_f32_16x16x32_bf16(false, a.b, false, bv.b, (short)0, acc, false, false);
  }
  for (int r = 0; r < 8; r++){
    int mr = m0 + r + 8*half;
    simsT[(long long)mr*64 + n0 + lan16] = acc[r];
  }
}

// ---------------- per-batch exact top-512 (desc value, ties by index asc) ----------------
__global__ void topk512(const float* __restrict__ simsT, int Nvocab, int* __restrict__ topIdx){
  __shared__ int redi[256];
  __shared__ unsigned keys[1024];
  __shared__ int idxs[1024];
  __shared__ unsigned s_cnt;
  int b = blockIdx.x, t = threadIdx.x;
  unsigned lo = 0u, hi = 0xFFFFFFFFu;
  for (int it = 0; it < 32; ++it){
    unsigned mid = (unsigned)(((unsigned long long)lo + (unsigned long long)hi + 1ull) >> 1);
    int c = 0;
    for (int v = t; v < Nvocab; v += 256)
      c += (fkey(simsT[(long long)v*64 + b]) >= mid) ? 1 : 0;
    redi[t] = c; __syncthreads();
    for (int s = 128; s > 0; s >>= 1){ if (t < s) redi[t] += redi[t + s]; __syncthreads(); }
    int cnt = redi[0]; __syncthreads();
    if (cnt >= 512) lo = mid; else hi = mid - 1;
  }
  if (t == 0) s_cnt = 0;
  for (int i = t; i < 1024; i += 256){ keys[i] = 0u; idxs[i] = 0x7FFFFFFF; }
  __syncthreads();
  for (int v = t; v < Nvocab; v += 256){
    unsigned k = fkey(simsT[(long long)v*64 + b]);
    if (k >= lo){
      unsigned p = atomicAdd(&s_cnt, 1u);
      if (p < 1024u){ keys[p] = k; idxs[p] = v; }
    }
  }
  __syncthreads();
  for (unsigned k = 2; k <= 1024; k <<= 1){
    for (unsigned j = k >> 1; j > 0; j >>= 1){
      for (unsigned i = t; i < 1024; i += 256){
        unsigned l = i ^ j;
        if (l > i){
          bool up = ((i & k) == 0);
          unsigned ka = keys[i], kb = keys[l]; int ia = idxs[i], ib = idxs[l];
          bool before = (ka > kb) || (ka == kb && ia < ib);
          if (up ? !before : before){ keys[i]=kb; keys[l]=ka; idxs[i]=ib; idxs[l]=ia; }
        }
      }
      __syncthreads();
    }
  }
  for (int i = t; i < 512; i += 256) topIdx[(long long)b*512 + i] = idxs[i];
}

// ---------------- combined scores + reward -> tot_reward(64,512) -------------------------
// block=64 (2 waves) per (batch, 16-candidate tile); wave w -> c-tile 16w
__global__ void combined_kernel(const float* __restrict__ vocab, const int* __restrict__ topIdx,
                                const unsigned short* __restrict__ Tn,
                                float* __restrict__ tot_reward){
  __shared__ __align__(32) unsigned short As[16][768];
  __shared__ float comb[16][32];
  int blk = blockIdx.x;
  int b = blk >> 5;
  int v0 = (blk & 31) * 16;
  int wave = threadIdx.x >> 5;
  int lane = threadIdx.x & 31;
  for (int j = 0; j < 8; j++){
    int row = wave*8 + j;
    int g = topIdx[(long long)b*512 + v0 + row];
    const float* src = vocab + (long long)g*768;
    float ss = 0.f;
    for (int d = lane; d < 768; d += 32){ float x = src[d]; ss += x*x; }
    for (int off = 16; off > 0; off >>= 1) ss += __shfl_xor(ss, off, 32);
    float inv = 1.0f / fmaxf(sqrtf(ss), EPSN);
    for (int d = lane; d < 768; d += 32) As[row][d] = f2bf(src[d]*inv);
  }
  __syncthreads();
  int half = lane >> 4, lan16 = lane & 15;
  int c0 = wave * 16;
  const unsigned short* brow = Tn + ((long long)b*32 + c0 + lan16)*768;
  v8f acc = {};
  for (int k0 = 0; k0 < 768; k0 += 32){
    BF16x16 a;
    *(v8s*)&a.u[0] = *(const v8s*)(&As[lan16][k0 + 8*half]);
    *(v8s*)&a.u[8] = *(const v8s*)(&As[lan16][k0 + 16 + 8*half]);
    BF16x16 bv;
    *(v8s*)&bv.u[0] = *(const v8s*)(brow + k0 + 8*half);
    *(v8s*)&bv.u[8] = *(const v8s*)(brow + k0 + 16 + 8*half);
    acc = __builtin_amdgcn_wmma_f32_16x16x32_bf16(false, a.b, false, bv.b, (short)0, acc, false, false);
  }
  for (int r = 0; r < 8; r++) comb[r + 8*half][c0 + lan16] = acc[r];
  __syncthreads();
  if (threadIdx.x < 16){
    int v = threadIdx.x;
    float maxNP = -1e30f; int cNP = 9;
    for (int c = 9; c < 25; ++c){ float x = comb[v][c]; if (x > maxNP){ maxNP = x; cNP = c; } }
    int primary = 0;
    for (int c = 0; c < 9; ++c){ float x = comb[v][c]; if (x > maxNP || (x == maxNP && c < cNP)) primary++; }
    float secondary = (cNP < 17) ? 0.0f : (cNP < 24 ? 1.0f : -10.0f);
    tot_reward[(long long)b*512 + v0 + v] = (float)primary + secondary;
  }
}

// ---------------- final: sort 512 rewards twice, pool top/bottom 256, search emb ----------
__global__ void final_kernel(const float* __restrict__ vocab, const int* __restrict__ topIdx,
                             const float* __restrict__ tot_reward,
                             float* __restrict__ out_search, float* __restrict__ out_maxp,
                             float* __restrict__ out_minp){
  __shared__ float vals[512];
  __shared__ int vid[512];
  __shared__ float red[256];
  int b = blockIdx.x, t = threadIdx.x;
  for (int i = t; i < 512; i += 256){ vals[i] = tot_reward[(long long)b*512 + i]; vid[i] = i; }
  __syncthreads();
  // sort: (value desc, idx asc) == jax top_k order
  for (unsigned k = 2; k <= 512; k <<= 1){
    for (unsigned j = k >> 1; j > 0; j >>= 1){
      for (unsigned i = t; i < 512; i += 256){
        unsigned l = i ^ j;
        if (l > i){
          bool up = ((i & k) == 0);
          float va = vals[i], vb = vals[l]; int ia = vid[i], ib = vid[l];
          bool before = (va > vb) || (va == vb && ia < ib);
          if (up ? !before : before){ vals[i]=vb; vals[l]=va; vid[i]=ib; vid[l]=ia; }
        }
      }
      __syncthreads();
    }
  }
  {
    int g = topIdx[(long long)b*512 + vid[0]];
    const float* src = vocab + (long long)g*768;
    for (int q = 0; q < 3; q++) out_search[(long long)b*768 + t + q*256] = src[t + q*256];
  }
  {
    float a0 = 0, a1 = 0, a2 = 0;
    for (int j = 0; j < 256; j++){
      int g = topIdx[(long long)b*512 + vid[j]];
      const float* src = vocab + (long long)g*768;
      a0 += src[t]; a1 += src[t + 256]; a2 += src[t + 512];
    }
    a0 *= (1.f/256.f); a1 *= (1.f/256.f); a2 *= (1.f/256.f);
    float tot = block_reduce_sum(a0*a0 + a1*a1 + a2*a2, red);
    float inv = 1.0f / fmaxf(sqrtf(tot), EPSN);
    out_maxp[(long long)b*768 + t]       = a0*inv;
    out_maxp[(long long)b*768 + t + 256] = a1*inv;
    out_maxp[(long long)b*768 + t + 512] = a2*inv;
  }
  __syncthreads();
  // re-sort: (value asc, idx asc) == top_k(-x) order
  for (unsigned k = 2; k <= 512; k <<= 1){
    for (unsigned j = k >> 1; j > 0; j >>= 1){
      for (unsigned i = t; i < 512; i += 256){
        unsigned l = i ^ j;
        if (l > i){
          bool up = ((i & k) == 0);
          float va = vals[i], vb = vals[l]; int ia = vid[i], ib = vid[l];
          bool before = (va < vb) || (va == vb && ia < ib);
          if (up ? !before : before){ vals[i]=vb; vals[l]=va; vid[i]=ib; vid[l]=ia; }
        }
      }
      __syncthreads();
    }
  }
  {
    float a0 = 0, a1 = 0, a2 = 0;
    for (int j = 0; j < 256; j++){
      int g = topIdx[(long long)b*512 + vid[j]];
      const float* src = vocab + (long long)g*768;
      a0 += src[t]; a1 += src[t + 256]; a2 += src[t + 512];
    }
    a0 *= (1.f/256.f); a1 *= (1.f/256.f); a2 *= (1.f/256.f);
    float tot = block_reduce_sum(a0*a0 + a1*a1 + a2*a2, red);
    float inv = 1.0f / fmaxf(sqrtf(tot), EPSN);
    out_minp[(long long)b*768 + t]       = a0*inv;
    out_minp[(long long)b*768 + t + 256] = a1*inv;
    out_minp[(long long)b*768 + t + 512] = a2*inv;
  }
}

extern "C" void kernel_launch(void* const* d_in, const int* in_sizes, int n_in,
                              void* d_out, int out_size, void* d_ws, size_t ws_size,
                              hipStream_t stream){
  (void)n_in; (void)out_size; (void)ws_size;
  const float* pos   = (const float*)d_in[0];
  const float* neg   = (const float*)d_in[1];
  const float* neut  = (const float*)d_in[2];
  const float* assas = (const float*)d_in[3];
  const float* vocab = (const float*)d_in[4];
  const float* W1 = (const float*)d_in[5];
  const float* b1 = (const float*)d_in[6];
  const float* W2 = (const float*)d_in[7];
  const float* b2 = (const float*)d_in[8];
  const float* W3 = (const float*)d_in[9];
  const float* b3 = (const float*)d_in[10];
  const float* W4 = (const float*)d_in[11];
  const float* b4 = (const float*)d_in[12];
  int Nvocab = in_sizes[4] / 768;

  char* ws = (char*)d_ws;
  size_t off = 0;
  auto alloc = [&](size_t bytes){ size_t p = off; off += (bytes + 255) & ~(size_t)255; return p; };
  unsigned short* WT1 = (unsigned short*)(ws + alloc((size_t)2304*3072*2));
  unsigned short* WT2 = (unsigned short*)(ws + alloc((size_t)1728*2304*2));
  unsigned short* WT3 = (unsigned short*)(ws + alloc((size_t)1024*1728*2));
  unsigned short* WT4 = (unsigned short*)(ws + alloc((size_t)768*1024*2));
  float* X  = (float*)(ws + alloc((size_t)64*3072*4));
  float* H1 = (float*)(ws + alloc((size_t)64*2304*4));
  float* H2 = (float*)(ws + alloc((size_t)64*1728*4));
  float* H3 = (float*)(ws + alloc((size_t)64*1024*4));
  float* H4 = (float*)(ws + alloc((size_t)64*768*4));
  unsigned short* MO_BF = (unsigned short*)(ws + alloc((size_t)64*768*2));
  unsigned short* TN_BF = (unsigned short*)(ws + alloc((size_t)64*32*768*2));
  float* SIMST = (float*)(ws + alloc((size_t)Nvocab*64*4));
  int*   TOPIDX = (int*)(ws + alloc((size_t)64*512*4));
  float* TOT = (float*)(ws + alloc((size_t)64*512*4));

  float* out = (float*)d_out;
  float* out_model  = out;
  float* out_search = out + (size_t)64*768;
  float* out_maxp   = out + (size_t)2*64*768;
  float* out_minp   = out + (size_t)3*64*768;

  // weight prep (independent)
  { long long tot = (long long)2304*3072; wt_prep<<<(unsigned)((tot+255)/256),256,0,stream>>>(W1,3072,2304,WT1,3072,2304); }
  { long long tot = (long long)1728*2304; wt_prep<<<(unsigned)((tot+255)/256),256,0,stream>>>(W2,2304,1700,WT2,2304,1728); }
  { long long tot = (long long)1024*1728; wt_prep<<<(unsigned)((tot+255)/256),256,0,stream>>>(W3,1700,1000,WT3,1728,1024); }
  { long long tot = (long long)768*1024;  wt_prep<<<(unsigned)((tot+255)/256),256,0,stream>>>(W4,1000,768, WT4,1024,768);  }

  pool_concat<<<64,256,0,stream>>>(pos, neg, neut, assas, X);
  tn_prep<<<64*32,256,0,stream>>>(pos, neg, neut, assas, TN_BF);

  // MLP chain
  gemm_mlp<<<2304/16,128,0,stream>>>(X,  3072, WT1, b1, 2304, H1, 2304, 1);
  gemm_mlp<<<1728/16,128,0,stream>>>(H1, 2304, WT2, b2, 1700, H2, 1728, 1);
  gemm_mlp<<<1024/16,128,0,stream>>>(H2, 1728, WT3, b3, 1000, H3, 1024, 1);
  gemm_mlp<<< 768/16,128,0,stream>>>(H3, 1024, WT4, b4,  768, H4,  768, 0);

  finalize_model_out<<<64,256,0,stream>>>(H4, out_model, MO_BF);

  // vocab similarity + selection
  sims_kernel<<<Nvocab/16,128,0,stream>>>(vocab, MO_BF, SIMST, Nvocab);
  topk512<<<64,256,0,stream>>>(SIMST, Nvocab, TOPIDX);

  // candidate scoring + rewards
  combined_kernel<<<64*32,64,0,stream>>>(vocab, TOPIDX, TN_BF, TOT);

  // final sorts + pooling
  final_kernel<<<64,256,0,stream>>>(vocab, TOPIDX, TOT, out_search, out_maxp, out_minp);
}
